// Model_17789754540486
// MI455X (gfx1250) — compile-verified, matrix operand order checked by reading, not run on hardware
//
#include <hip/hip_runtime.h>

// ---------------------------------------------------------------------------
// Problem constants
// ---------------------------------------------------------------------------
#define BATCH  4
#define SEQ    2048
#define DIN    4096
#define DOUT   1024
#define MTOT   (BATCH * SEQ)          // 8192
#define QK_SCALE 0.03125f             // 1/sqrt(1024)

// Tile config: 256 threads = 8 waves (wave32), wave grid 4(M) x 2(N).
// Block tile 128(M) x 128(N) x 32(K); each wave owns 2x4 tiles of 16x16.
// Double-buffered LDS: 2 * (128*40 + 128*40) * 2B = 40 KB.
#define BM 128
#define BN 128
#define BK 32
#define LP 40                          // LDS pitch in shorts (80B, 16B aligned)

typedef __attribute__((ext_vector_type(16))) __bf16 v16bf;
typedef __attribute__((ext_vector_type(8)))  float  v8f;
typedef __attribute__((__vector_size__(16))) int    vi4;   // matches builtin param

union BFrag { v16bf v; uint4 q[2]; unsigned short h[16]; };
union H8    { uint4 q; unsigned short h[8]; };

typedef unsigned short LdsTile[BM][LP];

// ---------------------------------------------------------------------------
// CDNA5 async global->LDS copy (bypasses VGPRs, tracked by ASYNCcnt).
// Guarded: falls back to sync VGPR staging if the builtins are absent.
// ---------------------------------------------------------------------------
#define __AS1 __attribute__((address_space(1)))
#define __AS3 __attribute__((address_space(3)))

#if defined(__gfx1250__) && __has_builtin(__builtin_amdgcn_global_load_async_to_lds_b128)
  #define HAS_ASYNC_LDS 1
  #define ASYNC_COPY16(gsrc, ldst)                                              \
      __builtin_amdgcn_global_load_async_to_lds_b128(                           \
          (__AS1 vi4*)(size_t)(gsrc), (__AS3 vi4*)(ldst), 0, 0)
  #if __has_builtin(__builtin_amdgcn_s_wait_asynccnt)
    #define ASYNC_WAIT(n) __builtin_amdgcn_s_wait_asynccnt((unsigned short)(n))
  #else
    #define ASYNC_WAIT(n) asm volatile("s_wait_asynccnt %0" :: "i"(n) : "memory")
  #endif
#else
  #define HAS_ASYNC_LDS 0
  #define ASYNC_COPY16(gsrc, ldst)                                              \
      (*reinterpret_cast<uint4*>(ldst) = *reinterpret_cast<const uint4*>(gsrc))
  #define ASYNC_WAIT(n)
#endif

__device__ __forceinline__ float bf16_to_f32(unsigned short h) {
    unsigned u = ((unsigned)h) << 16;
    float f; __builtin_memcpy(&f, &u, 4);
    return f;
}
__device__ __forceinline__ unsigned short f32_to_bf16(float f) {
    unsigned u; __builtin_memcpy(&u, &f, 4);
    u += 0x7fffu + ((u >> 16) & 1u);
    return (unsigned short)(u >> 16);
}
__device__ __forceinline__ unsigned mix32(unsigned x) {
    x ^= x >> 16; x *= 0x7feb352du;
    x ^= x >> 15; x *= 0x846ca68bu;
    x ^= x >> 16;
    return x;
}

// ---------------------------------------------------------------------------
// Unified GEMM building blocks.  A: [M][K] bf16 rows.  B: Bt = [N][K] bf16 rows
// (i.e. B pre-transposed in memory) -> all staging is contiguous b128 chunks.
// ---------------------------------------------------------------------------
__device__ __forceinline__ void stage_tile(
    const unsigned short* __restrict__ g, int ld, int r0, int k0,
    unsigned short (&t)[BM][LP], int tid)
{
    #pragma unroll
    for (int i = 0; i < 2; ++i) {                 // 512 chunks, 2 per thread
        int chunk = tid + i * 256;
        int row = chunk >> 2;                     // 4 chunks per 32-wide row
        int c8  = (chunk & 3) * 8;
        const unsigned short* src = g + (size_t)(r0 + row) * ld + k0 + c8;
        ASYNC_COPY16(src, &t[row][c8]);
    }
}

__device__ __forceinline__ void compute_tile(
    const unsigned short (&ta)[BM][LP], const unsigned short (&tb)[BM][LP],
    v8f acc[2][4], int wm, int wn, int l16, int half)
{
    BFrag a[2], b[4];
    #pragma unroll
    for (int mi = 0; mi < 2; ++mi) {
        int row = wm * 32 + mi * 16 + l16;
        a[mi].q[0] = *reinterpret_cast<const uint4*>(&ta[row][half * 8]);
        a[mi].q[1] = *reinterpret_cast<const uint4*>(&ta[row][16 + half * 8]);
    }
    #pragma unroll
    for (int ni = 0; ni < 4; ++ni) {
        int col = wn * 64 + ni * 16 + l16;
        b[ni].q[0] = *reinterpret_cast<const uint4*>(&tb[col][half * 8]);
        b[ni].q[1] = *reinterpret_cast<const uint4*>(&tb[col][16 + half * 8]);
    }
    #pragma unroll
    for (int mi = 0; mi < 2; ++mi)
        #pragma unroll
        for (int ni = 0; ni < 4; ++ni)
            acc[mi][ni] = __builtin_amdgcn_wmma_f32_16x16x32_bf16(
                false, a[mi].v, false, b[ni].v, (short)0, acc[mi][ni], false, false);
}

// Full double-buffered K loop (async prefetch one stage ahead).
__device__ __forceinline__ void gemm_loop(
    const unsigned short* __restrict__ A, int lda, int m0,
    const unsigned short* __restrict__ Bt, int ldb, int n0, int K,
    LdsTile* sa, LdsTile* sb, v8f acc[2][4],
    int tid, int wm, int wn, int l16, int half)
{
    stage_tile(A,  lda, m0, 0, sa[0], tid);
    stage_tile(Bt, ldb, n0, 0, sb[0], tid);
    int buf = 0;
    for (int k0 = 0; k0 < K; k0 += BK) {
        if (k0 + BK < K) {
            stage_tile(A,  lda, m0, k0 + BK, sa[buf ^ 1], tid);
            stage_tile(Bt, ldb, n0, k0 + BK, sb[buf ^ 1], tid);
            ASYNC_WAIT(4);            // 4 async insts outstanding = the prefetch
        } else {
            ASYNC_WAIT(0);
        }
        __syncthreads();
        compute_tile(sa[buf], sb[buf], acc, wm, wn, l16, half);
        __syncthreads();
        buf ^= 1;
    }
}

// ---------------------------------------------------------------------------
// Kernel 0a: x1 f32 -> bf16
// ---------------------------------------------------------------------------
__global__ __launch_bounds__(256) void convert_x(
    const float* __restrict__ X, unsigned short* __restrict__ XB)
{
    size_t base = ((size_t)blockIdx.x * 256 + threadIdx.x) * 8;
    float4 f0 = *reinterpret_cast<const float4*>(&X[base]);
    float4 f1 = *reinterpret_cast<const float4*>(&X[base + 4]);
    H8 o;
    o.h[0] = f32_to_bf16(f0.x); o.h[1] = f32_to_bf16(f0.y);
    o.h[2] = f32_to_bf16(f0.z); o.h[3] = f32_to_bf16(f0.w);
    o.h[4] = f32_to_bf16(f1.x); o.h[5] = f32_to_bf16(f1.y);
    o.h[6] = f32_to_bf16(f1.z); o.h[7] = f32_to_bf16(f1.w);
    *reinterpret_cast<uint4*>(&XB[base]) = o.q;
}

// ---------------------------------------------------------------------------
// Kernel 0b: W[z] f32 [K=4096][N=1024] -> WT bf16 [z][N][K]  (tiled via LDS)
// ---------------------------------------------------------------------------
__global__ __launch_bounds__(256) void transpose_w(
    const float* __restrict__ Wq, const float* __restrict__ Wk,
    const float* __restrict__ Wv, unsigned short* __restrict__ WT)
{
    const int z = blockIdx.z;
    const float* W = (z == 0) ? Wq : (z == 1 ? Wk : Wv);
    unsigned short* Out = WT + (size_t)z * DOUT * DIN;

    __shared__ unsigned short t[64][72];          // [n][k] bf16, padded
    const int tid = threadIdx.x;
    const int k0 = blockIdx.x * 64;
    const int n0 = blockIdx.y * 64;

    #pragma unroll
    for (int i = 0; i < 4; ++i) {                 // 64x64 f32 = 1024 float4
        int chunk = tid + i * 256;
        int kr = chunk >> 4;                      // 16 float4 per 64-wide row
        int c4 = (chunk & 15) * 4;
        float4 f = *reinterpret_cast<const float4*>(&W[(size_t)(k0 + kr) * DOUT + n0 + c4]);
        t[c4 + 0][kr] = f32_to_bf16(f.x);
        t[c4 + 1][kr] = f32_to_bf16(f.y);
        t[c4 + 2][kr] = f32_to_bf16(f.z);
        t[c4 + 3][kr] = f32_to_bf16(f.w);
    }
    __syncthreads();
    #pragma unroll
    for (int i = 0; i < 2; ++i) {                 // 64 rows x 8 b128 chunks
        int chunk = tid + i * 256;
        int nr = chunk >> 3;
        int c8 = (chunk & 7) * 8;
        *reinterpret_cast<uint4*>(&Out[(size_t)(n0 + nr) * DIN + k0 + c8]) =
            *reinterpret_cast<const uint4*>(&t[nr][c8]);
    }
}

// ---------------------------------------------------------------------------
// Kernel 1: QKV GEMM.  XB[8192,4096] x WT[z] -> Q,K bf16 [M][1024] (+bias);
// V written TRANSPOSED per batch: VT[b][d][t]  (packed b128 stores!)
// ---------------------------------------------------------------------------
__global__ __launch_bounds__(256) void qkv_gemm(
    const unsigned short* __restrict__ XB, const unsigned short* __restrict__ WT,
    const float* __restrict__ bq, const float* __restrict__ bk,
    const float* __restrict__ bv,
    unsigned short* __restrict__ Q, unsigned short* __restrict__ Kc,
    unsigned short* __restrict__ VT)
{
    const int z = blockIdx.z;
    const unsigned short* Bt = WT + (size_t)z * DOUT * DIN;
    const float* bias = (z == 0) ? bq : (z == 1 ? bk : bv);

    __shared__ LdsTile sa[2];
    __shared__ LdsTile sb[2];

    const int tid = threadIdx.x, lane = tid & 31, wave = tid >> 5;
    const int wm = wave & 3, wn = wave >> 2;
    const int l16 = lane & 15, half = lane >> 4;
    const int m0 = blockIdx.x * BM, n0 = blockIdx.y * BN;

    v8f acc[2][4] = {};
    gemm_loop(XB, DIN, m0, Bt, DIN, n0, DIN, sa, sb, acc, tid, wm, wn, l16, half);

    #pragma unroll
    for (int mi = 0; mi < 2; ++mi) {
        #pragma unroll
        for (int ni = 0; ni < 4; ++ni) {
            int col = n0 + wn * 64 + ni * 16 + l16;
            float bb = bias[col];
            int rbase = m0 + wm * 32 + mi * 16 + half * 8;
            if (z < 2) {
                unsigned short* Out = (z == 0) ? Q : Kc;
                #pragma unroll
                for (int i = 0; i < 8; ++i)
                    Out[(size_t)(rbase + i) * DOUT + col] = f32_to_bf16(acc[mi][ni][i] + bb);
            } else {
                int b = rbase >> 11, r = rbase & 2047;    // batch, row-in-batch
                H8 o;
                #pragma unroll
                for (int i = 0; i < 8; ++i) o.h[i] = f32_to_bf16(acc[mi][ni][i] + bb);
                *reinterpret_cast<uint4*>(&VT[((size_t)b * DOUT + col) * SEQ + r]) = o.q;
            }
        }
    }
}

// ---------------------------------------------------------------------------
// Kernel 2: per-batch Q x K^T -> attn scores bf16 (K rows are already [N][K]).
// ---------------------------------------------------------------------------
__global__ __launch_bounds__(256) void qk_gemm(
    const unsigned short* __restrict__ Q, const unsigned short* __restrict__ K,
    unsigned short* __restrict__ Attn)
{
    const int batch = blockIdx.z;
    const unsigned short* Qb = Q + (size_t)batch * SEQ * DOUT;
    const unsigned short* Kb = K + (size_t)batch * SEQ * DOUT;
    unsigned short* Ab = Attn + (size_t)batch * SEQ * SEQ;

    __shared__ LdsTile sa[2];
    __shared__ LdsTile sb[2];

    const int tid = threadIdx.x, lane = tid & 31, wave = tid >> 5;
    const int wm = wave & 3, wn = wave >> 2;
    const int l16 = lane & 15, half = lane >> 4;
    const int m0 = blockIdx.x * BM, n0 = blockIdx.y * BN;

    v8f acc[2][4] = {};
    gemm_loop(Qb, DOUT, m0, Kb, DOUT, n0, DOUT, sa, sb, acc, tid, wm, wn, l16, half);

    #pragma unroll
    for (int mi = 0; mi < 2; ++mi) {
        #pragma unroll
        for (int ni = 0; ni < 4; ++ni) {
            int col = n0 + wn * 64 + ni * 16 + l16;
            int rbase = m0 + wm * 32 + mi * 16 + half * 8;
            #pragma unroll
            for (int i = 0; i < 8; ++i)
                Ab[(size_t)(rbase + i) * SEQ + col] = f32_to_bf16(acc[mi][ni][i] * QK_SCALE);
        }
    }
}

// ---------------------------------------------------------------------------
// Kernel 3: row softmax + deterministic dropout (p=0.5), in-place bf16.
// ---------------------------------------------------------------------------
__global__ __launch_bounds__(256) void softmax_dropout(unsigned short* __restrict__ Attn)
{
    const size_t row = blockIdx.x;
    unsigned short* rp = Attn + row * SEQ;
    const int tid = threadIdx.x;
    __shared__ float red[256];

    H8 in;
    in.q = *reinterpret_cast<const uint4*>(&rp[tid * 8]);
    float v[8];
    #pragma unroll
    for (int j = 0; j < 8; ++j) v[j] = bf16_to_f32(in.h[j]);

    float m = v[0];
    #pragma unroll
    for (int j = 1; j < 8; ++j) m = fmaxf(m, v[j]);
    red[tid] = m;
    __syncthreads();
    for (int s = 128; s > 0; s >>= 1) {
        if (tid < s) red[tid] = fmaxf(red[tid], red[tid + s]);
        __syncthreads();
    }
    m = red[0];
    __syncthreads();

    float lsum = 0.f;
    #pragma unroll
    for (int j = 0; j < 8; ++j) { v[j] = __expf(v[j] - m); lsum += v[j]; }
    red[tid] = lsum;
    __syncthreads();
    for (int s = 128; s > 0; s >>= 1) {
        if (tid < s) red[tid] += red[tid + s];
        __syncthreads();
    }
    const float inv = 1.0f / red[0];

    H8 outp;
    #pragma unroll
    for (int j = 0; j < 8; ++j) {
        unsigned gidx = (unsigned)(row * SEQ) + (unsigned)(tid * 8 + j);
        unsigned h = mix32(gidx ^ 0x2A6B1E35u);
        float p = (h & 1u) ? (v[j] * inv * 2.0f) : 0.0f;
        outp.h[j] = f32_to_bf16(p);
    }
    *reinterpret_cast<uint4*>(&rp[tid * 8]) = outp.q;
}

// ---------------------------------------------------------------------------
// Kernel 4: per-batch attn[2048,2048] x V -> out f32.  B = VT rows (direct).
// ---------------------------------------------------------------------------
__global__ __launch_bounds__(256) void av_gemm(
    const unsigned short* __restrict__ Attn, const unsigned short* __restrict__ VT,
    float* __restrict__ Out)
{
    const int batch = blockIdx.z;
    const unsigned short* Ab = Attn + (size_t)batch * SEQ * SEQ;
    const unsigned short* Vt = VT + (size_t)batch * DOUT * SEQ;   // [d][t]
    float* Ob = Out + (size_t)batch * SEQ * DOUT;

    __shared__ LdsTile sa[2];
    __shared__ LdsTile sb[2];

    const int tid = threadIdx.x, lane = tid & 31, wave = tid >> 5;
    const int wm = wave & 3, wn = wave >> 2;
    const int l16 = lane & 15, half = lane >> 4;
    const int m0 = blockIdx.x * BM, n0 = blockIdx.y * BN;

    v8f acc[2][4] = {};
    gemm_loop(Ab, SEQ, m0, Vt, SEQ, n0, SEQ, sa, sb, acc, tid, wm, wn, l16, half);

    #pragma unroll
    for (int mi = 0; mi < 2; ++mi) {
        #pragma unroll
        for (int ni = 0; ni < 4; ++ni) {
            int col = n0 + wn * 64 + ni * 16 + l16;
            int rbase = m0 + wm * 32 + mi * 16 + half * 8;
            #pragma unroll
            for (int i = 0; i < 8; ++i)
                Ob[(size_t)(rbase + i) * DOUT + col] = acc[mi][ni][i];
        }
    }
}

// ---------------------------------------------------------------------------
// Launch
// ---------------------------------------------------------------------------
extern "C" void kernel_launch(void* const* d_in, const int* in_sizes, int n_in,
                              void* d_out, int out_size, void* d_ws, size_t ws_size,
                              hipStream_t stream)
{
    (void)in_sizes; (void)n_in; (void)out_size; (void)ws_size;

    const float* x1 = (const float*)d_in[0];
    const float* Wq = (const float*)d_in[1];
    const float* bq = (const float*)d_in[2];
    const float* Wk = (const float*)d_in[3];
    const float* bk = (const float*)d_in[4];
    const float* Wv = (const float*)d_in[5];
    const float* bv = (const float*)d_in[6];
    float* out = (float*)d_out;

    // Workspace (bf16 as ushort):
    //   xb   [8192][4096]          67.1 MB
    //   wT   [3][1024][4096]       25.2 MB
    //   q,k  [8192][1024] each     16.8 MB each
    //   vT   [4][1024][2048]       16.8 MB
    //   attn [4][2048][2048]       33.6 MB
    unsigned short* xb   = (unsigned short*)d_ws;
    unsigned short* wT   = xb + (size_t)MTOT * DIN;
    unsigned short* qb   = wT + (size_t)3 * DOUT * DIN;
    unsigned short* kbuf = qb + (size_t)MTOT * DOUT;
    unsigned short* vT   = kbuf + (size_t)MTOT * DOUT;
    unsigned short* attn = vT + (size_t)MTOT * DOUT;

    dim3 blk(256);

    convert_x<<<dim3((MTOT * (size_t)DIN) / (256 * 8)), blk, 0, stream>>>(x1, xb);
    transpose_w<<<dim3(DIN / 64, DOUT / 64, 3), blk, 0, stream>>>(Wq, Wk, Wv, wT);
    qkv_gemm<<<dim3(MTOT / BM, DOUT / BN, 3), blk, 0, stream>>>(
        xb, wT, bq, bk, bv, qb, kbuf, vT);
    qk_gemm<<<dim3(SEQ / BM, SEQ / BN, BATCH), blk, 0, stream>>>(qb, kbuf, attn);
    softmax_dropout<<<dim3(MTOT), blk, 0, stream>>>(attn);
    av_gemm<<<dim3(SEQ / BM, DOUT / BN, BATCH), blk, 0, stream>>>(attn, vT, out);
}
